// mySeq2SeqModel_31018253812271
// MI455X (gfx1250) — matrix operand order repaired and checked by reading, not compiled, not run on hardware
//
#include <hip/hip_runtime.h>

// ---- CDNA5 / gfx1250 seq2seq RNN, single fused persistent kernel ----
// V=27, E=64, H=128, B=512, T=256.  32 workgroups x 256 threads (8 wave32).
// Latency-bound 512-step serial recurrence; batch rows independent -> each WG
// owns 16 rows for the whole sequence. Weights f16 in registers (B-fragments),
// double-buffered h state f16 in LDS (bank-conflict-free padded stride, one
// barrier per step), per-step GEMMs via v_wmma_f32_16x16x32_f16 in two
// independent accumulation chains (hides WMMA latency), hw tanh.

#define VV 27
#define EE 64
#define HH 128
#define BB 512
#define TT 256

#define HSTRIDE 136   // h row stride in halves: 68 DWORDs -> bank shift 4/row
#define ESTRIDE 72    // emb row stride in halves: 36 DWORDs -> bank spread
#define HBUF    (16 * HSTRIDE)   // 2176 halves per h buffer

// LDS layout (halves). sU/sW are transient (register fragments loaded once per
// phase), so sWo reuses the sU region in phase 1.
#define OFF_U   0       // U^T   [n=128][k=128]   16384
#define OFF_W   16384   // W^T   [n=128][k=64]     8192
#define OFF_WO  0       // Wout^T[n=32 ][k=128]    4096 (overlaps dead sU)
#define OFF_EMB 24576   // emb   [27][ESTRIDE]     1944
#define OFF_H   26520   // h     2 x [16][HSTRIDE] 4352 (ping-pong)
#define LDS_HALVES (OFF_H + 2 * HBUF)   // 30872 halves = ~60 KB

typedef _Float16 h8   __attribute__((ext_vector_type(8)));
typedef _Float16 v16h __attribute__((ext_vector_type(16)));
typedef float    v8f  __attribute__((ext_vector_type(8)));

// A-matrix (16x32 f16, MxK) fragment: lane l (m=l&15, hi=l>>4)
// holds k in [K0+16*hi, +16): two contiguous 16B runs (matches B fragments).
static __device__ __forceinline__ v16h wmma_ld_a(const _Float16* row, int K0, int hi) {
  h8 lo = *(const h8*)(row + K0 + 16 * hi);
  h8 hh = *(const h8*)(row + K0 + 16 * hi + 8);
  return __builtin_shufflevector(lo, hh, 0, 1, 2, 3, 4, 5, 6, 7,
                                         8, 9, 10, 11, 12, 13, 14, 15);
}

static __device__ __forceinline__ v8f wmma_mac(v16h a, v16h b, v8f c) {
  return __builtin_amdgcn_wmma_f32_16x16x32_f16(false, a, false, b,
                                                (short)0, c, false, false);
}

static __device__ __forceinline__ void sched_fence() {
#if __has_builtin(__builtin_amdgcn_sched_barrier)
  __builtin_amdgcn_sched_barrier(0);   // keep load group ahead of compute
#endif
}

// Branchless tanh on the v_exp_f32 / v_rcp_f32 hardware paths:
// tanh(x) = 1 - 2/(e^{2x}+1); clamp keeps exp2 in range (tanh(10)==1.0f).
static __device__ __forceinline__ float fast_tanh(float x) {
  float xc = fminf(fmaxf(x, -10.0f), 10.0f);
#if __has_builtin(__builtin_amdgcn_exp2f)
  float t = __builtin_amdgcn_exp2f(xc * 2.8853900817779268f);  // 2*log2(e)
#else
  float t = exp2f(xc * 2.8853900817779268f);
#endif
#if __has_builtin(__builtin_amdgcn_rcpf)
  float r = __builtin_amdgcn_rcpf(t + 1.0f);
#else
  float r = 1.0f / (t + 1.0f);
#endif
  return fmaf(-2.0f, r, 1.0f);
}

__global__ __launch_bounds__(256)
void rnn_seq2seq(const int* __restrict__ enc_ids, const int* __restrict__ dec_ids,
                 const float* __restrict__ emb,
                 const float* __restrict__ W_enc, const float* __restrict__ U_enc,
                 const float* __restrict__ b_enc,
                 const float* __restrict__ W_dec, const float* __restrict__ U_dec,
                 const float* __restrict__ b_dec,
                 const float* __restrict__ W_out, const float* __restrict__ b_out,
                 float* __restrict__ out)
{
  extern __shared__ _Float16 smem[];
  _Float16* sU   = smem + OFF_U;
  _Float16* sW   = smem + OFF_W;
  _Float16* sWo  = smem + OFF_WO;
  _Float16* sEmb = smem + OFF_EMB;
  _Float16* sH   = smem + OFF_H;    // two ping-pong h buffers

  const int tid  = threadIdx.x;
  const int wave = tid >> 5;        // 0..7 (wave32)
  const int lane = tid & 31;
  const int hi   = lane >> 4;
  const int lx   = lane & 15;
  const int b0   = blockIdx.x * 16; // this WG's 16 batch rows
  const int ncol = wave * 16 + lx;  // B-matrix column owned by this lane

  // ---- one-time LDS fills (emb table + h0 = 0 in buffer 0) ----
  for (int idx = tid; idx < VV * EE; idx += 256) {
    int v = idx >> 6, e = idx & (EE - 1);
    sEmb[v * ESTRIDE + e] = (_Float16)emb[idx];
  }
  for (int idx = tid; idx < HBUF; idx += 256) sH[idx] = (_Float16)0.0f;

  for (int phase = 0; phase < 2; ++phase) {
    const int*   ids = phase ? dec_ids : enc_ids;
    const float* Wg  = phase ? W_dec : W_enc;
    const float* Ug  = phase ? U_dec : U_enc;
    const float* bg  = phase ? b_dec : b_enc;

    // stage this phase's weights as f16, transposed for B-fragment loads
    for (int idx = tid; idx < HH * HH; idx += 256) {
      int k = idx >> 7, n = idx & (HH - 1);
      sU[n * HH + k] = (_Float16)Ug[idx];
    }
    for (int idx = tid; idx < EE * HH; idx += 256) {
      int k = idx >> 7, n = idx & (HH - 1);
      sW[n * EE + k] = (_Float16)Wg[idx];
    }
    __syncthreads();

    // persistent B fragments in registers (column ncol; 16 contiguous halves
    // at k = kt*32 + 16*hi  -> 2x ds_load_b128 each)
    v16h Uf[4], Wf[2];
#pragma unroll
    for (int kt = 0; kt < 4; ++kt)
      Uf[kt] = *(const v16h*)(sU + ncol * HH + kt * 32 + 16 * hi);
#pragma unroll
    for (int kt = 0; kt < 2; ++kt)
      Wf[kt] = *(const v16h*)(sW + ncol * EE + kt * 32 + 16 * hi);
    const float bn = bg[ncol];

    __syncthreads();  // all fragment reads done; sU region is now dead
    if (phase == 1) { // stage Wout^T into the dead sU region
      for (int idx = tid; idx < 32 * HH; idx += 256) {
        int n = idx >> 7, k = idx & (HH - 1);
        sWo[idx] = (n < VV) ? (_Float16)W_out[k * VV + n] : (_Float16)0.0f;
      }
    }
    __syncthreads();

    v16h Wof[4];
    float bo = 0.0f;
    if (phase == 1 && wave < 2) {   // waves 0-1 own the logits tiles
#pragma unroll
      for (int kt = 0; kt < 4; ++kt)
        Wof[kt] = *(const v16h*)(sWo + ncol * HH + kt * 32 + 16 * hi);
      if (ncol < VV) bo = b_out[ncol];
    }

    // software-pipelined token-id stream for this lane's batch row
    const int* idrow = ids + (b0 + lx) * TT;
    int id_cur = idrow[0];

    for (int t = 0; t < TT; ++t) {
      const int tn = (t + 1 < TT) ? (t + 1) : t;
      const int id_nxt = idrow[tn];          // lands one full step later

      // ping-pong h buffers: read parity t, write parity t^1.
      // (TT even -> encoder's final h ends in buffer 0 for the decoder.)
      const _Float16* hrd = sH + ((t & 1) ? HBUF : 0);
      _Float16*       hwr = sH + ((t & 1) ? 0 : HBUF);

      // gather all A fragments up front (independent, conflict-free ds loads)
      const _Float16* erow = sEmb + id_cur * ESTRIDE;
      const _Float16* hrow = hrd + lx * HSTRIDE;
      v16h Ae0 = wmma_ld_a(erow, 0, hi);
      v16h Ae1 = wmma_ld_a(erow, 32, hi);
      v16h Ah0 = wmma_ld_a(hrow, 0, hi);
      v16h Ah1 = wmma_ld_a(hrow, 32, hi);
      v16h Ah2 = wmma_ld_a(hrow, 64, hi);
      v16h Ah3 = wmma_ld_a(hrow, 96, hi);
      sched_fence();                   // don't sink loads into the WMMA chain

      // two independent 3-deep accumulation chains -> WMMAs interleave
      v8f c0, c1 = {};
#pragma unroll
      for (int r = 0; r < 8; ++r) c0[r] = bn;  // bias folded into chain 0
      c0 = wmma_mac(Ae0, Wf[0], c0);   // x @ W   (E = 64 -> 2 k-tiles)
      c1 = wmma_mac(Ah1, Uf[1], c1);
      c0 = wmma_mac(Ae1, Wf[1], c0);
      c1 = wmma_mac(Ah2, Uf[2], c1);
      c0 = wmma_mac(Ah0, Uf[0], c0);   // h @ U   (H = 128 -> 4 k-tiles)
      c1 = wmma_mac(Ah3, Uf[3], c1);

      // h = tanh(c0+c1); C layout: lane l, vgpr r -> [m = r + 8*hi][n = ncol]
#pragma unroll
      for (int r = 0; r < 8; ++r) {
        float hval = fast_tanh(c0[r] + c1[r]);
        hwr[(r + 8 * hi) * HSTRIDE + ncol] = (_Float16)hval;
      }
      __syncthreads();   // single barrier/step: new h visible to everyone

      // fused output projection: logits_t = h_t @ Wout + b_out
      if (phase == 1 && wave < 2) {
        const _Float16* hnew = hwr + lx * HSTRIDE;
        v16h al0 = wmma_ld_a(hnew, 0, hi);
        v16h al1 = wmma_ld_a(hnew, 32, hi);
        v16h al2 = wmma_ld_a(hnew, 64, hi);
        v16h al3 = wmma_ld_a(hnew, 96, hi);
        v8f cl0, cl1 = {};
#pragma unroll
        for (int r = 0; r < 8; ++r) cl0[r] = bo;
        cl0 = wmma_mac(al0, Wof[0], cl0);
        cl1 = wmma_mac(al2, Wof[2], cl1);
        cl0 = wmma_mac(al1, Wof[1], cl0);
        cl1 = wmma_mac(al3, Wof[3], cl1);
        if (ncol < VV) {
#pragma unroll
          for (int r = 0; r < 8; ++r) {
            int m = r + 8 * hi;
            __builtin_nontemporal_store(cl0[r] + cl1[r],
                out + ((b0 + m) * TT + t) * VV + ncol);
          }
        }
      }
      id_cur = id_nxt;
    }
    __syncthreads();                   // before next phase re-stages weights
  }
}

extern "C" void kernel_launch(void* const* d_in, const int* in_sizes, int n_in,
                              void* d_out, int out_size, void* d_ws, size_t ws_size,
                              hipStream_t stream) {
  (void)in_sizes; (void)n_in; (void)out_size; (void)d_ws; (void)ws_size;
  const int*   enc_ids = (const int*)d_in[0];
  const int*   dec_ids = (const int*)d_in[1];
  const float* emb     = (const float*)d_in[2];
  const float* W_enc   = (const float*)d_in[3];
  const float* U_enc   = (const float*)d_in[4];
  const float* b_enc   = (const float*)d_in[5];
  const float* W_dec   = (const float*)d_in[6];
  const float* U_dec   = (const float*)d_in[7];
  const float* b_dec   = (const float*)d_in[8];
  const float* W_out   = (const float*)d_in[9];
  const float* b_out   = (const float*)d_in[10];
  float* out = (float*)d_out;

  const size_t shmem = LDS_HALVES * 2;  // ~60 KB dynamic LDS (of 320 KB/WGP)
  rnn_seq2seq<<<dim3(BB / 16), dim3(256), shmem, stream>>>(
      enc_ids, dec_ids, emb, W_enc, U_enc, b_enc,
      W_dec, U_dec, b_dec, W_out, b_out, out);
}
